// CrossWindowAttention_67422396613083
// MI455X (gfx1250) — compile-verified, matrix operand order checked by reading, not run on hardware
//
#include <hip/hip_runtime.h>
#include <hip/hip_bf16.h>
#include <hip/hip_fp16.h>

typedef __attribute__((ext_vector_type(16))) _Float16 v16h;
typedef __attribute__((ext_vector_type(8)))  float    v8f;

#define DIMC 256
#define HEADS 8
#define HD 32
#define MTOK 64
#define NTOK 65
#define NWIN 64
#define BIMG 32
#define BW (BIMG * NWIN)      /* 2048 window-batches */
#define ROWS (BW * NTOK)      /* 133120 token rows (divisible by 16) */
#define NPAD 80               /* token dim padded to 5 x 16 tiles   */
#define KPAD 96               /* softmax K dim padded to 3 x 32     */

// 16-bit A/B fragment element map (ISA 7.12.2): half j of lane holds
// K = (j>>3)*16 + (lane>=16 ? 8 : 0) + (j&7)  -> two contiguous 8-half runs,
// which the compiler lowers to paired ds_load_b128.
__device__ __forceinline__ int kmap(int j, int lane) {
  return ((j >> 3) << 4) + ((lane & 16) >> 1) + (j & 7);
}

// ---------------------------------------------------------------------------
// GEMM: C[r][c] = (sum_k A[r][k] * W[c][k] + bias[c]) * scale, K = 256 fixed.
// One wave32 per 16x64 output tile: the A fragment is reused across 4 column
// tiles (4 WMMAs per A-fragment load). A/W tiles are contiguous in memory
// (K == row pitch) so LDS staging is fully coalesced.
// ---------------------------------------------------------------------------
template<bool A_F16, bool OUT_F32>
__global__ void __launch_bounds__(32)
gemm16x64_wmma(const void* __restrict__ A_, const float* __restrict__ W,
               const float* __restrict__ bias, void* __restrict__ C_,
               int cols, float scale) {
  __shared__ _Float16 As[16 * DIMC];   // 8 KB
  __shared__ _Float16 Ws[64 * DIMC];   // 32 KB
  const int lane = threadIdx.x;
  const int rowTile = blockIdx.x;
  const int colTile = blockIdx.y;      // 64-column tiles

  {
    const size_t abase = (size_t)rowTile * 16 * DIMC;
    const size_t wbase = (size_t)colTile * 64 * DIMC;
    if (A_F16) {
      const _Float16* Ah = (const _Float16*)A_;
      for (int idx = lane; idx < 16 * DIMC; idx += 32) As[idx] = Ah[abase + idx];
    } else {
      const float* Af = (const float*)A_;
      for (int idx = lane; idx < 16 * DIMC; idx += 32) As[idx] = (_Float16)Af[abase + idx];
    }
    for (int idx = lane; idx < 64 * DIMC; idx += 32) Ws[idx] = (_Float16)W[wbase + idx];
  }
  __syncthreads();

  v8f acc0 = {}, acc1 = {}, acc2 = {}, acc3 = {};
  const int r = lane & 15;
#pragma unroll
  for (int k0 = 0; k0 < DIMC; k0 += 32) {
    v16h a;
#pragma unroll
    for (int j = 0; j < 16; ++j) a[j] = As[r * DIMC + k0 + kmap(j, lane)];

    v16h b0, b1, b2, b3;
#pragma unroll
    for (int j = 0; j < 16; ++j) {
      const int kk = k0 + kmap(j, lane);
      b0[j] = Ws[(0 * 16 + r) * DIMC + kk];
      b1[j] = Ws[(1 * 16 + r) * DIMC + kk];
      b2[j] = Ws[(2 * 16 + r) * DIMC + kk];
      b3[j] = Ws[(3 * 16 + r) * DIMC + kk];
    }
    acc0 = __builtin_amdgcn_wmma_f32_16x16x32_f16(false, a, false, b0, (short)0, acc0, false, false);
    acc1 = __builtin_amdgcn_wmma_f32_16x16x32_f16(false, a, false, b1, (short)0, acc1, false, false);
    acc2 = __builtin_amdgcn_wmma_f32_16x16x32_f16(false, a, false, b2, (short)0, acc2, false, false);
    acc3 = __builtin_amdgcn_wmma_f32_16x16x32_f16(false, a, false, b3, (short)0, acc3, false, false);
  }

  const int orow0 = rowTile * 16 + ((lane >> 4) << 3);
  v8f accs[4] = {acc0, acc1, acc2, acc3};
#pragma unroll
  for (int ct = 0; ct < 4; ++ct) {
    const int ocol = colTile * 64 + ct * 16 + r;
    const float bv = bias[ocol];
#pragma unroll
    for (int rr = 0; rr < 8; ++rr) {
      const float v = (accs[ct][rr] + bv) * scale;
      const size_t oi = (size_t)(orow0 + rr) * (size_t)cols + ocol;
      if (OUT_F32) ((float*)C_)[oi] = v;
      else         ((_Float16*)C_)[oi] = (_Float16)v;
    }
  }
}

// ---------------------------------------------------------------------------
// Padded relative-position bias: biasPad[h][r][c], r,c in [0,65), zero for
// the prefix-token row/col. 135 KB -> lives in L2 for the attention pass.
// ---------------------------------------------------------------------------
__global__ void bias_gather(const float* __restrict__ bias_table,
                            const int* __restrict__ rel_index,
                            float* __restrict__ biasPad) {
  const int idx = blockIdx.x * blockDim.x + threadIdx.x;
  const int tot = HEADS * NTOK * NTOK;
  if (idx >= tot) return;
  const int h = idx / (NTOK * NTOK);
  const int rem = idx % (NTOK * NTOK);
  const int r = rem / NTOK, c = rem % NTOK;
  float v = 0.f;
  if (r > 0 && c > 0)
    v = bias_table[rel_index[(r - 1) * MTOK + (c - 1)] * HEADS + h];
  biasPad[idx] = v;
}

// ---------------------------------------------------------------------------
// Fused attention, one wave32 per (window-batch b, head h).
// S = q k^T (WMMA, K=32 == head dim), bias+mask+softmax in LDS, O = P v.
// S never touches HBM (would be ~1.4 GB round-trip otherwise).
// ---------------------------------------------------------------------------
__global__ void __launch_bounds__(32)
attn_wmma(const _Float16* __restrict__ qh, const _Float16* __restrict__ kvh,
          const float* __restrict__ biasPad, const float* __restrict__ mask,
          _Float16* __restrict__ aout) {
  __shared__ _Float16 qs[NPAD * HD];
  __shared__ _Float16 ks[NPAD * HD];
  __shared__ _Float16 vs[KPAD * HD];
  __shared__ float    Ssm[NPAD * NPAD];

  const int lane = threadIdx.x;
  const int b = blockIdx.x >> 3;
  const int h = blockIdx.x & 7;
  const int w = b & (NWIN - 1);

  // ---- stage q/k/v rows (zero padding rows) -------------------------------
  for (int n = lane; n < KPAD; n += 32) {
    uint32_t* dv = (uint32_t*)(vs + n * HD);
    if (n < NPAD) {
      uint32_t* dq = (uint32_t*)(qs + n * HD);
      uint32_t* dk = (uint32_t*)(ks + n * HD);
      if (n < NTOK) {
        const uint32_t* sq = (const uint32_t*)(qh  + (size_t)(b * NTOK + n) * DIMC + h * HD);
        const uint32_t* sk = (const uint32_t*)(kvh + (size_t)(b * NTOK + n) * (2 * DIMC) + h * HD);
        const uint32_t* sv = (const uint32_t*)(kvh + (size_t)(b * NTOK + n) * (2 * DIMC) + DIMC + h * HD);
#pragma unroll
        for (int j = 0; j < 16; ++j) { dq[j] = sq[j]; dk[j] = sk[j]; dv[j] = sv[j]; }
      } else {
#pragma unroll
        for (int j = 0; j < 16; ++j) { dq[j] = 0u; dk[j] = 0u; dv[j] = 0u; }
      }
    } else {
#pragma unroll
      for (int j = 0; j < 16; ++j) dv[j] = 0u;
    }
  }
  __syncthreads();

  // ---- S = q k^T : 5x5 tiles, one WMMA per tile (K = 32 = head dim) -------
#pragma unroll
  for (int i = 0; i < 5; ++i) {
    v16h a;
#pragma unroll
    for (int j = 0; j < 16; ++j)
      a[j] = qs[(i * 16 + (lane & 15)) * HD + kmap(j, lane)];
#pragma unroll
    for (int jt = 0; jt < 5; ++jt) {
      v16h bf;
#pragma unroll
      for (int j = 0; j < 16; ++j)
        bf[j] = ks[(jt * 16 + (lane & 15)) * HD + kmap(j, lane)];
      v8f c = {};
      c = __builtin_amdgcn_wmma_f32_16x16x32_f16(false, a, false, bf, (short)0,
                                                 c, false, false);
      const int rbase = i * 16 + ((lane >> 4) << 3);
      const int cc = jt * 16 + (lane & 15);
#pragma unroll
      for (int rr = 0; rr < 8; ++rr)
        Ssm[(rbase + rr) * NPAD + cc] = c[rr];
    }
  }
  __syncthreads();

  // ---- bias + mask + softmax, in place; zero padding rows/cols ------------
  for (int row = lane; row < NPAD; row += 32) {
    float* srow = Ssm + row * NPAD;
    if (row < NTOK) {
      const float* brow = biasPad + ((size_t)h * NTOK + row) * NTOK;
      float m = -1e30f;
      for (int c = 0; c < NTOK; ++c) {
        const float mv = (row > 0 && c > 0)
                       ? mask[((size_t)w * MTOK + (row - 1)) * MTOK + (c - 1)] : 0.f;
        const float v = srow[c] + brow[c] + mv;
        srow[c] = v;
        m = fmaxf(m, v);
      }
      float s = 0.f;
      for (int c = 0; c < NTOK; ++c) { const float e = __expf(srow[c] - m); srow[c] = e; s += e; }
      const float inv = 1.f / s;
      for (int c = 0; c < NTOK; ++c) srow[c] *= inv;
      for (int c = NTOK; c < NPAD; ++c) srow[c] = 0.f;
    } else {
      for (int c = 0; c < NPAD; ++c) srow[c] = 0.f;
    }
  }
  __syncthreads();

  // ---- O = P v : 5 row tiles x 2 d tiles, K padded to 96 (3 WMMA steps) ---
#pragma unroll
  for (int i = 0; i < 5; ++i) {
#pragma unroll
    for (int nt = 0; nt < 2; ++nt) {
      v8f acc = {};
#pragma unroll
      for (int kt = 0; kt < 3; ++kt) {
        v16h a, bf;
#pragma unroll
        for (int j = 0; j < 16; ++j) {
          const int col = kt * 32 + kmap(j, lane);
          a[j]  = (col < NPAD) ? (_Float16)Ssm[(i * 16 + (lane & 15)) * NPAD + col]
                               : (_Float16)0.f;
          bf[j] = vs[col * HD + nt * 16 + (lane & 15)];
        }
        acc = __builtin_amdgcn_wmma_f32_16x16x32_f16(false, a, false, bf, (short)0,
                                                     acc, false, false);
      }
      const int rbase = i * 16 + ((lane >> 4) << 3);
      const int dd = nt * 16 + (lane & 15);
#pragma unroll
      for (int rr = 0; rr < 8; ++rr) {
        const int row = rbase + rr;
        if (row < NTOK)
          aout[(size_t)(b * NTOK + row) * DIMC + h * HD + dd] = (_Float16)acc[rr];
      }
    }
  }
}

extern "C" void kernel_launch(void* const* d_in, const int* in_sizes, int n_in,
                              void* d_out, int out_size, void* d_ws, size_t ws_size,
                              hipStream_t stream) {
  const float* x     = (const float*)d_in[0];
  const float* kv    = (const float*)d_in[1];
  const float* mask  = (const float*)d_in[2];
  const float* Wq    = (const float*)d_in[3];
  const float* bq    = (const float*)d_in[4];
  const float* Wkv   = (const float*)d_in[5];
  const float* bkv   = (const float*)d_in[6];
  const float* Wproj = (const float*)d_in[7];
  const float* bproj = (const float*)d_in[8];
  const float* btab  = (const float*)d_in[9];
  const int*   relix = (const int*)d_in[10];

  // workspace layout (f16 intermediates halve HBM traffic)
  _Float16* qh    = (_Float16*)d_ws;                       // ROWS*256 f16
  _Float16* kvh   = qh  + (size_t)ROWS * DIMC;             // ROWS*512 f16
  _Float16* aoutW = kvh + (size_t)ROWS * 2 * DIMC;         // ROWS*256 f16
  float*  biasPad = (float*)(aoutW + (size_t)ROWS * DIMC); // 8*65*65 f32

  const float scale = 0.17677669529663687f; // 32^-0.5, fused into Q proj

  hipLaunchKernelGGL((gemm16x64_wmma<false, false>),
                     dim3(ROWS / 16, DIMC / 64), dim3(32), 0, stream,
                     (const void*)x, Wq, bq, (void*)qh, DIMC, scale);
  hipLaunchKernelGGL((gemm16x64_wmma<false, false>),
                     dim3(ROWS / 16, (2 * DIMC) / 64), dim3(32), 0, stream,
                     (const void*)kv, Wkv, bkv, (void*)kvh, 2 * DIMC, 1.0f);
  bias_gather<<<(HEADS * NTOK * NTOK + 255) / 256, 256, 0, stream>>>(btab, relix, biasPad);
  attn_wmma<<<BW * HEADS, 32, 0, stream>>>(qh, kvh, biasPad, mask, aoutW);
  hipLaunchKernelGGL((gemm16x64_wmma<true, true>),
                     dim3(ROWS / 16, DIMC / 64), dim3(32), 0, stream,
                     (const void*)aoutW, Wproj, bproj, d_out, DIMC, 1.0f);
}
